// MambaModel2_39333310496909
// MI455X (gfx1250) — matrix operand order repaired and checked by reading, not compiled, not run on hardware
//
#include <hip/hip_runtime.h>
#include <math.h>

// ---- Model dimensions (fixed by the reference) ----
#define NL       24
#define DMODEL   768
#define ED       1536
#define NSTATE   16
#define KCONV    4
#define DT_RANK  48
#define VOCAB    1024
#define LSEQ     512
#define BATCH    2
#define BL       (BATCH * LSEQ)         // 1024 tokens
#define SP_COLS  (DT_RANK + 2 * NSTATE) // 80

typedef __attribute__((ext_vector_type(16))) __bf16 v16bf;
typedef __attribute__((ext_vector_type(8)))  float  v8f;

// ---------------------------------------------------------------------------
// Device helpers
// ---------------------------------------------------------------------------
__device__ __forceinline__ float silu_f(float x) {
    return x / (1.0f + __expf(-x));
}
__device__ __forceinline__ float softplus_f(float x) {
    return (x > 20.0f) ? x : log1pf(__expf(x));
}

// ---------------------------------------------------------------------------
// Embedding gather: h[b*L + l, :] = embed[src[l*B + b], :]
// ---------------------------------------------------------------------------
__global__ void embed_kernel(const int* __restrict__ src,
                             const float* __restrict__ embed,
                             float* __restrict__ h) {
    int m = blockIdx.x;               // token index, m = b*LSEQ + l
    int b = m / LSEQ;
    int l = m - b * LSEQ;
    int id = src[l * BATCH + b];
    const float* erow = embed + (size_t)id * DMODEL;
    float* hrow = h + (size_t)m * DMODEL;
    for (int i = threadIdx.x; i < DMODEL; i += blockDim.x) hrow[i] = erow[i];
}

// ---------------------------------------------------------------------------
// RMSNorm: out[row,:] = x[row,:] * rsqrt(mean(x^2)+eps) * w[:]
// ---------------------------------------------------------------------------
__global__ void rmsnorm_kernel(const float* __restrict__ x,
                               const float* __restrict__ w,
                               float* __restrict__ out, int d) {
    __shared__ float red[256];
    int row = blockIdx.x, tid = threadIdx.x;
    const float* xr = x + (size_t)row * d;
    float s = 0.0f;
    for (int i = tid; i < d; i += 256) { float v = xr[i]; s += v * v; }
    red[tid] = s;
    __syncthreads();
    for (int st = 128; st > 0; st >>= 1) {
        if (tid < st) red[tid] += red[tid + st];
        __syncthreads();
    }
    float rstd = rsqrtf(red[0] / (float)d + 1e-5f);
    float* orow = out + (size_t)row * d;
    for (int i = tid; i < d; i += 256) orow[i] = xr[i] * rstd * w[i];
}

// ---------------------------------------------------------------------------
// WMMA GEMM:  C[m,n] = sum_k A[m,k] * W[n,k]  (fp32 memory, bf16 WMMA,
// fp32 accumulate).  One wave computes a 16 x (16*TN) strip: the A fragment
// is loaded once per K-step and reused across TN back-to-back
// v_wmma_f32_16x16x32_bf16 ops.
//
// KDIM/LDA/LDB are compile-time so the in-bounds predicates fold away
// (KDIM % 32 == 0) or the 2-iteration K=48 loop fully unrolls -> branchless
// batched loads, no exec-mask churn around the WMMAs.
//
// MODE 0: C = acc
// MODE 1: C = softplus(acc + bias[n])
// MODE 2: C += acc                       (residual, in-place)
// MODE 3: C[(l*BATCH+b)*ldc + n] = acc   (logits transpose, m = b*LSEQ + l)
// ---------------------------------------------------------------------------
template <int MODE, int KDIM, int LDA, int LDB, int TN>
__global__ void gemm_wmma_bf16(const float* __restrict__ A,
                               const float* __restrict__ W,
                               float* __restrict__ C, int ldc,
                               int M, int Ndim,
                               const float* __restrict__ bias) {
    int wave = (int)((blockIdx.x * blockDim.x + threadIdx.x) >> 5);
    int lane = threadIdx.x & 31;
    int ngrp = Ndim / (16 * TN);          // N-tile groups
    int tiles_m = M >> 4;
    if (wave >= tiles_m * ngrp) return;   // uniform per-wave
    int tm = wave / ngrp;
    int tg = wave - tm * ngrp;

    int hi = (lane & 16) ? 1 : 0;
    int rowA  = tm * 16 + (lane & 15);
    int colB0 = tg * 16 * TN + (lane & 15);
    const float* arow = A + (size_t)rowA * LDA;

    v8f acc[TN];
#pragma unroll
    for (int j = 0; j < TN; ++j) acc[j] = {};

#pragma unroll
    for (int kk = 0; kk < KDIM; kk += 32) {
        // ---- A fragment (16x32, MxK):
        //      k = 2*(i&3) + (i>=4 ? 16 : 0) + (hi ? 8 : 0)
        v16bf af;
#pragma unroll
        for (int i = 0; i < 8; ++i) {
            int ka = kk + 2 * (i & 3) + ((i & 4) ? 16 : 0) + (hi ? 8 : 0);
            bool inb = (KDIM % 32 == 0) ? true : (ka < KDIM);
            float2 av = inb ? *reinterpret_cast<const float2*>(arow + ka)
                            : make_float2(0.0f, 0.0f);
            af[2 * i]     = (__bf16)av.x;
            af[2 * i + 1] = (__bf16)av.y;
        }
        // ---- TN x { B fragment (32x16, KxN): k = 2*(i&3) + (i>=4 ? 16 : 0) + (hi ? 8 : 0) ; wmma }
#pragma unroll
        for (int j = 0; j < TN; ++j) {
            const float* brow = W + (size_t)(colB0 + j * 16) * LDB;
            v16bf bf;
#pragma unroll
            for (int i = 0; i < 8; ++i) {
                int kb = kk + 2 * (i & 3) + ((i & 4) ? 16 : 0) + (hi ? 8 : 0);
                bool inb = (KDIM % 32 == 0) ? true : (kb < KDIM);
                float2 bv = inb ? *reinterpret_cast<const float2*>(brow + kb)
                                : make_float2(0.0f, 0.0f);
                bf[2 * i]     = (__bf16)bv.x;
                bf[2 * i + 1] = (__bf16)bv.y;
            }
            acc[j] = __builtin_amdgcn_wmma_f32_16x16x32_bf16(
                /*neg_a=*/false, af, /*neg_b=*/false, bf,
                /*c_mod=*/(short)0, acc[j], /*reuse_a=*/false, /*reuse_b=*/false);
        }
    }

    // Epilogue. C/D layout: n = lane&15 ; m = vgpr + 8*hi
#pragma unroll
    for (int j = 0; j < TN; ++j) {
        int n = tg * 16 * TN + j * 16 + (lane & 15);
#pragma unroll
        for (int r = 0; r < 8; ++r) {
            int m = tm * 16 + r + 8 * hi;
            float v = acc[j][r];
            if (MODE == 0) {
                C[(size_t)m * ldc + n] = v;
            } else if (MODE == 1) {
                C[(size_t)m * ldc + n] = softplus_f(v + bias[n]);
            } else if (MODE == 2) {
                C[(size_t)m * ldc + n] += v;
            } else {  // MODE 3: transpose [B,L,V] -> [L,B,V]
                int b = m / LSEQ, l = m - b * LSEQ;
                C[((size_t)l * BATCH + b) * ldc + n] = v;
            }
        }
    }
}

// ---------------------------------------------------------------------------
// Depthwise causal conv1d (K=4) + bias + SiLU.
// ---------------------------------------------------------------------------
__global__ void conv_silu_kernel(const float* __restrict__ ug,
                                 const float* __restrict__ cw,
                                 const float* __restrict__ cb,
                                 float* __restrict__ uc) {
    long long idx = (long long)blockIdx.x * blockDim.x + threadIdx.x;
    if (idx >= (long long)BATCH * LSEQ * ED) return;
    int e = (int)(idx % ED);
    int t = (int)((idx / ED) % LSEQ);
    int b = (int)(idx / ((long long)ED * LSEQ));
    float s = cb[e];
#pragma unroll
    for (int k = 0; k < KCONV; ++k) {
        int tt = t + k - (KCONV - 1);
        if (tt >= 0)
            s += ug[((size_t)(b * LSEQ + tt)) * (2 * ED) + e] * cw[e * KCONV + k];
    }
    uc[((size_t)(b * LSEQ + t)) * ED + e] = silu_f(s);
}

// ---------------------------------------------------------------------------
// Selective scan: one thread per (b,e) channel, 16 states in VGPRs,
// B_t/C_t staged through LDS; fused D skip + SiLU gate.
// ---------------------------------------------------------------------------
__global__ void scan_kernel(const float* __restrict__ sp,    // [BL, 80]
                            const float* __restrict__ dtp,   // [BL, ED]
                            const float* __restrict__ uc,    // [BL, ED]
                            const float* __restrict__ ug,    // [BL, 2ED]
                            const float* __restrict__ A_log, // [ED, N]
                            const float* __restrict__ Dp,    // [ED]
                            float* __restrict__ y) {         // [BL, ED]
    __shared__ float Bs[NSTATE];
    __shared__ float Cs[NSTATE];
    int e = blockIdx.x * blockDim.x + threadIdx.x;
    int b = blockIdx.y;

    float Arow[NSTATE], hst[NSTATE];
#pragma unroll
    for (int n = 0; n < NSTATE; ++n) {
        Arow[n] = -__expf(A_log[(size_t)e * NSTATE + n]);
        hst[n] = 0.0f;
    }
    float dskip = Dp[e];

    for (int t = 0; t < LSEQ; ++t) {
        int m = b * LSEQ + t;
        if (threadIdx.x < 2 * NSTATE) {
            int n = threadIdx.x;
            float v = sp[(size_t)m * SP_COLS + DT_RANK + n];
            if (n < NSTATE) Bs[n] = v; else Cs[n - NSTATE] = v;
        }
        __syncthreads();

        float dtv = dtp[(size_t)m * ED + e];
        float u   = uc[(size_t)m * ED + e];
        float du  = dtv * u;
        float yv  = 0.0f;
#pragma unroll
        for (int n = 0; n < NSTATE; ++n) {
            float dA = __expf(dtv * Arow[n]);
            hst[n] = dA * hst[n] + du * Bs[n];
            yv += hst[n] * Cs[n];
        }
        yv += u * dskip;
        float g = ug[(size_t)m * (2 * ED) + ED + e];
        yv *= silu_f(g);
        y[(size_t)m * ED + e] = yv;
        __syncthreads();   // protect Bs/Cs before next step
    }
}

// ---------------------------------------------------------------------------
// Host side
// ---------------------------------------------------------------------------
static inline int gemm_blocks(int M, int Ndim, int TN) {
    int waves = (M >> 4) * (Ndim / (16 * TN));
    return (waves * 32 + 255) / 256;
}

extern "C" void kernel_launch(void* const* d_in, const int* in_sizes, int n_in,
                              void* d_out, int out_size, void* d_ws, size_t ws_size,
                              hipStream_t stream) {
    (void)in_sizes; (void)n_in; (void)out_size; (void)ws_size;

    const int*   src      = (const int*)  d_in[0];
    const float* embed    = (const float*)d_in[1];
    const float* norm_w   = (const float*)d_in[2];
    const float* in_proj  = (const float*)d_in[3];
    const float* conv_w   = (const float*)d_in[4];
    const float* conv_b   = (const float*)d_in[5];
    const float* x_proj   = (const float*)d_in[6];
    const float* dt_w     = (const float*)d_in[7];
    const float* dt_b     = (const float*)d_in[8];
    const float* A_log    = (const float*)d_in[9];
    const float* Dp       = (const float*)d_in[10];
    const float* out_proj = (const float*)d_in[11];
    const float* normf_w  = (const float*)d_in[12];
    float* out = (float*)d_out;

    // Workspace layout (fp32)
    float* ws  = (float*)d_ws;
    float* h   = ws;                         // [BL, D]
    float* xn  = h   + (size_t)BL * DMODEL;  // [BL, D]
    float* ug  = xn  + (size_t)BL * DMODEL;  // [BL, 2ED]
    float* uc  = ug  + (size_t)BL * 2 * ED;  // [BL, ED]
    float* sp  = uc  + (size_t)BL * ED;      // [BL, 80]
    float* dtp = sp  + (size_t)BL * SP_COLS; // [BL, ED]
    float* yb  = dtp + (size_t)BL * ED;      // [BL, ED]

    // 1) embedding
    embed_kernel<<<BL, 256, 0, stream>>>(src, embed, h);

    // 2) layers
    for (int layer = 0; layer < NL; ++layer) {
        const float* nw   = norm_w   + (size_t)layer * DMODEL;
        const float* Wi   = in_proj  + (size_t)layer * 2 * ED * DMODEL;
        const float* cw   = conv_w   + (size_t)layer * ED * KCONV;
        const float* cb   = conv_b   + (size_t)layer * ED;
        const float* Wx   = x_proj   + (size_t)layer * SP_COLS * ED;
        const float* Wdt  = dt_w     + (size_t)layer * ED * DT_RANK;
        const float* bdt  = dt_b     + (size_t)layer * ED;
        const float* Alog = A_log    + (size_t)layer * ED * NSTATE;
        const float* dpp  = Dp       + (size_t)layer * ED;
        const float* Wo   = out_proj + (size_t)layer * DMODEL * ED;

        // RMSNorm
        rmsnorm_kernel<<<BL, 256, 0, stream>>>(h, nw, xn, DMODEL);

        // ug = xn @ W_in^T   [BL x 2ED], K = D
        gemm_wmma_bf16<0, DMODEL, DMODEL, DMODEL, 4>
            <<<gemm_blocks(BL, 2 * ED, 4), 256, 0, stream>>>(
                xn, Wi, ug, 2 * ED, BL, 2 * ED, nullptr);

        // causal depthwise conv + SiLU -> uc
        {
            long long tot = (long long)BATCH * LSEQ * ED;
            int blocks = (int)((tot + 255) / 256);
            conv_silu_kernel<<<blocks, 256, 0, stream>>>(ug, cw, cb, uc);
        }

        // sp = uc @ W_x^T    [BL x 80], K = ED  (5 N-tiles -> TN=5)
        gemm_wmma_bf16<0, ED, ED, ED, 5>
            <<<gemm_blocks(BL, SP_COLS, 5), 256, 0, stream>>>(
                uc, Wx, sp, SP_COLS, BL, SP_COLS, nullptr);

        // dtp = softplus(sp[:, :48] @ W_dt^T + b_dt)   [BL x ED], K = 48
        gemm_wmma_bf16<1, DT_RANK, SP_COLS, DT_RANK, 4>
            <<<gemm_blocks(BL, ED, 4), 256, 0, stream>>>(
                sp, Wdt, dtp, ED, BL, ED, bdt);

        // selective scan + D skip + SiLU gate -> yb
        {
            dim3 grid(ED / 256, BATCH);
            scan_kernel<<<grid, 256, 0, stream>>>(sp, dtp, uc, ug, Alog, dpp, yb);
        }

        // h += yb @ W_out^T  (residual, in-place), K = ED
        gemm_wmma_bf16<2, ED, ED, ED, 4>
            <<<gemm_blocks(BL, DMODEL, 4), 256, 0, stream>>>(
                yb, Wo, h, DMODEL, BL, DMODEL, nullptr);
    }

    // 3) final norm + tied lm head, transposed write [L, B, V]
    rmsnorm_kernel<<<BL, 256, 0, stream>>>(h, normf_w, xn, DMODEL);
    gemm_wmma_bf16<3, DMODEL, DMODEL, DMODEL, 4>
        <<<gemm_blocks(BL, VOCAB, 4), 256, 0, stream>>>(
            xn, embed, out, VOCAB, BL, VOCAB, nullptr);
}
